// PlaneDCN_80642305950165
// MI455X (gfx1250) — matrix operand order repaired and checked by reading, not compiled, not run on hardware
//
#include <hip/hip_runtime.h>
#include <hip/hip_bf16.h>

// ---- problem constants (match reference) ----
#define BB   4
#define CINC 64
#define COUTC 64
#define HC   128
#define WC   160
#define HWC  (HC * WC)          // 20480
#define CMIN 72                 // 64 + 8 similarity channels
#define CMINP 96                // padded to 3 * 32 for WMMA K-steps
#define CMOUT 27
#define TILES_PER_B 1280        // (H*W)/16
#define WTILES 10               // W/16

typedef __bf16 bf16_t;
typedef __attribute__((ext_vector_type(16))) __bf16 v16bf;
typedef __attribute__((ext_vector_type(8)))  __bf16 v8bf;
typedef __attribute__((ext_vector_type(8)))  float  v8f;

__device__ __forceinline__ unsigned short f2bf(float f) {
  union { float f; unsigned int u; } cv; cv.f = f;
  unsigned int u = cv.u;
  unsigned int r = u + 0x7FFFu + ((u >> 16) & 1u);   // round-to-nearest-even
  return (unsigned short)(r >> 16);
}

__device__ __forceinline__ v16bf cat16(v8bf lo, v8bf hi) {
  return __builtin_shufflevector(lo, hi, 0,1,2,3,4,5,6,7,8,9,10,11,12,13,14,15);
}

// ===================================================================
// Kernel 1: per-pixel cosine similarity (8 channels) -> ws
// ===================================================================
__global__ void __launch_bounds__(256)
sim_kernel(const float* __restrict__ x, float* __restrict__ sim)
{
  const int gid = blockIdx.x * 256 + threadIdx.x;   // one pixel per thread
  const int b   = gid / HWC;
  const int rem = gid % HWC;
  const int h   = rem / WC;
  const int w   = rem % WC;

  float dot[9], nrm[9];
  #pragma unroll
  for (int k = 0; k < 9; ++k) { dot[k] = 0.f; nrm[k] = 0.f; }
  float ncc = 0.f;

  const float* xb = x + (size_t)b * CINC * HWC;
  for (int c = 0; c < CINC; ++c) {
    const float* xc = xb + c * HWC;
    if (c + 1 < CINC) __builtin_prefetch(xc + HWC + rem, 0, 3);  // next channel plane
    const float ctr = xc[rem];
    ncc += ctr * ctr;
    #pragma unroll
    for (int k = 0; k < 9; ++k) {
      const int hh = h + k / 3 - 1;
      const int w2 = w + k % 3 - 1;
      float v = 0.f;
      if (hh >= 0 && hh < HC && w2 >= 0 && w2 < WC) v = xc[hh * WC + w2];
      dot[k] += ctr * v;
      nrm[k] += v * v;
    }
  }
  const float nc = fmaxf(sqrtf(ncc), 1e-8f);
  #pragma unroll
  for (int k = 0; k < 9; ++k) {
    if (k == 4) continue;                       // skip center tap
    const float cosv = dot[k] / (nc * fmaxf(sqrtf(nrm[k]), 1e-8f));
    const int j = (k < 4) ? k : k - 1;
    sim[((b * 8 + j) * HWC) + rem] = cosv;
  }
}

// ===================================================================
// Kernel 2: 3x3 conv over 72ch (x ++ sim) -> 27ch raw "om" via WMMA
// Implicit GEMM: M=16-pixel tile, N=32 (27 used), K=9 taps * 96 (72 padded)
// ===================================================================
__global__ void __launch_bounds__(128)
cm_conv_kernel(const float* __restrict__ x, const float* __restrict__ sim,
               const float* __restrict__ cw, const float* __restrict__ cb,
               float* __restrict__ om)
{
  __shared__ alignas(16) unsigned short Bt[32][CMINP];       // [n][c] for one tap
  __shared__ alignas(16) unsigned short At[4][16][CMINP];    // per-wave A tile

  const int tid  = threadIdx.x;
  const int lane = tid & 31;
  const int wv   = tid >> 5;
  const int t    = blockIdx.x * 4 + wv;     // M-tile id
  const int b    = t / TILES_PER_B;
  const int r0   = t % TILES_PER_B;
  const int h    = r0 / WTILES;
  const int w0   = (r0 % WTILES) * 16;

  const int half = lane >> 4;
  const int mn   = lane & 15;

  v8f acc0 = {}; v8f acc1 = {};

  for (int tap = 0; tap < 9; ++tap) {
    const int ky = tap / 3 - 1;
    const int kx = tap % 3 - 1;
    __syncthreads();
    // stage B (block-wide): Bt[n][c] = cm_weight[n][c][tap], zero-padded
    for (int i = tid; i < 32 * CMINP; i += 128) {
      const int n = i / CMINP, c = i % CMINP;
      float v = 0.f;
      if (n < CMOUT && c < CMIN) v = cw[(n * CMIN + c) * 9 + tap];
      Bt[n][c] = f2bf(v);
    }
    // stage A (per-wave): 16 pixels x 96 channels (x then sim, zero pad)
    const int hh = h + ky;
    for (int i = lane; i < 16 * CMINP; i += 32) {
      const int p = i & 15, c = i >> 4;
      const int ww = w0 + p + kx;
      float v = 0.f;
      if (hh >= 0 && hh < HC && ww >= 0 && ww < WC && c < CMIN) {
        v = (c < CINC) ? x[((b * CINC + c) * HC + hh) * WC + ww]
                       : sim[((b * 8 + (c - CINC)) * HC + hh) * WC + ww];
      }
      At[wv][p][c] = f2bf(v);
    }
    // prefetch next tap's weight slice while WMMA runs
    if (tap < 8 && lane < 27)
      __builtin_prefetch(&cw[(lane * CMIN) * 9 + tap + 1], 0, 3);
    __syncthreads();
    #pragma unroll
    for (int ks = 0; ks < 3; ++ks) {
      const int ca = ks * 32 + half * 8;              // A: K 0-7/16-23 vs 8-15/24-31
      v8bf aLo = *(const v8bf*)&At[wv][mn][ca];
      v8bf aHi = *(const v8bf*)&At[wv][mn][ca + 16];
      v16bf av = cat16(aLo, aHi);
      const int cbb = ks * 32 + half * 8;             // B: K 0-7/16-23 vs 8-15/24-31
      v8bf b0l = *(const v8bf*)&Bt[mn][cbb];
      v8bf b0h = *(const v8bf*)&Bt[mn][cbb + 16];
      acc0 = __builtin_amdgcn_wmma_f32_16x16x32_bf16(
          false, av, false, cat16(b0l, b0h), (short)0, acc0, false, false);
      v8bf b1l = *(const v8bf*)&Bt[16 + mn][cbb];
      v8bf b1h = *(const v8bf*)&Bt[16 + mn][cbb + 16];
      acc1 = __builtin_amdgcn_wmma_f32_16x16x32_bf16(
          false, av, false, cat16(b1l, b1h), (short)0, acc1, false, false);
    }
  }

  // epilogue: lane holds N=mn, rows M = r + 8*half -> 8 consecutive pixels.
  // pixbase is 8-float (32B) aligned -> two b128 stores per output channel.
  const int pixbase = h * WC + w0 + half * 8;
  {
    const float bb = cb[mn];
    float4 lo = make_float4(acc0[0] + bb, acc0[1] + bb, acc0[2] + bb, acc0[3] + bb);
    float4 hi = make_float4(acc0[4] + bb, acc0[5] + bb, acc0[6] + bb, acc0[7] + bb);
    float* dst = &om[((b * CMOUT + mn) * HWC) + pixbase];
    *(float4*)(dst)     = lo;
    *(float4*)(dst + 4) = hi;
  }
  if (16 + mn < CMOUT) {
    const float bb = cb[16 + mn];
    float4 lo = make_float4(acc1[0] + bb, acc1[1] + bb, acc1[2] + bb, acc1[3] + bb);
    float4 hi = make_float4(acc1[4] + bb, acc1[5] + bb, acc1[6] + bb, acc1[7] + bb);
    float* dst = &om[((b * CMOUT + 16 + mn) * HWC) + pixbase];
    *(float4*)(dst)     = lo;
    *(float4*)(dst + 4) = hi;
  }
}

// ===================================================================
// Kernel 3: deformable conv via WMMA
// per tap: bilinear-gather 16px x 64ch (mask & validity folded into
// the 4 interp weights) -> bf16 A tile; weight tap-slice -> B tile;
// K = 9 taps * 64 ch (2 K-steps each), N = 64 (4 N-tiles)
// ===================================================================
__global__ void __launch_bounds__(128)
dcn_kernel(const float* __restrict__ x, const float* __restrict__ om,
           const float* __restrict__ w, const float* __restrict__ bias,
           float* __restrict__ out)
{
  __shared__ alignas(16) unsigned short Bt[64][64];     // [n][c] for one tap
  __shared__ alignas(16) unsigned short At[4][16][64];  // per-wave A tile
  __shared__ float pw[4][16][4];                        // interp weights (mask folded)
  __shared__ int   pi[4][16][4];                        // clamped plane indices

  const int tid  = threadIdx.x;
  const int lane = tid & 31;
  const int wv   = tid >> 5;
  const int t    = blockIdx.x * 4 + wv;
  const int b    = t / TILES_PER_B;
  const int r0   = t % TILES_PER_B;
  const int h    = r0 / WTILES;
  const int w0   = (r0 % WTILES) * 16;

  const int half = lane >> 4;
  const int mn   = lane & 15;

  v8f acc[4] = {{}, {}, {}, {}};

  for (int tap = 0; tap < 9; ++tap) {
    __syncthreads();
    // stage B (block-wide): Bt[o][c] = weight[o][c][tap]
    for (int i = tid; i < 64 * 64; i += 128) {
      const int n = i >> 6, c = i & 63;
      Bt[n][c] = f2bf(w[(n * CINC + c) * 9 + tap]);
    }
    // per-(pixel, tap) bilinear params (lanes 0..15)
    if (lane < 16) {
      const int p = lane;
      const int wp = w0 + p;
      const int pix = h * WC + wp;
      const float dy = om[((b * CMOUT + 2 * tap) * HWC) + pix];
      const float dx = om[((b * CMOUT + 2 * tap + 1) * HWC) + pix];
      const float mv = om[((b * CMOUT + 18 + tap) * HWC) + pix];
      const float msk = 1.f / (1.f + __expf(-mv));
      const float py = dy + (float)(h + tap / 3 - 1);
      const float px = dx + (float)(wp + tap % 3 - 1);
      const float fy = floorf(py), fx = floorf(px);
      const int y0 = (int)fy, x0 = (int)fx;
      const float wy = py - fy, wx = px - fx;
      const int y1 = y0 + 1, x1 = x0 + 1;
      const bool vy0 = (y0 >= 0) && (y0 < HC);
      const bool vy1 = (y1 >= 0) && (y1 < HC);
      const bool vx0 = (x0 >= 0) && (x0 < WC);
      const bool vx1 = (x1 >= 0) && (x1 < WC);
      const int y0c = min(max(y0, 0), HC - 1);
      const int y1c = min(max(y1, 0), HC - 1);
      const int x0c = min(max(x0, 0), WC - 1);
      const int x1c = min(max(x1, 0), WC - 1);
      pi[wv][p][0] = y0c * WC + x0c;
      pi[wv][p][1] = y0c * WC + x1c;
      pi[wv][p][2] = y1c * WC + x0c;
      pi[wv][p][3] = y1c * WC + x1c;
      pw[wv][p][0] = (vy0 && vx0) ? (1.f - wy) * (1.f - wx) * msk : 0.f;
      pw[wv][p][1] = (vy0 && vx1) ? (1.f - wy) * wx * msk : 0.f;
      pw[wv][p][2] = (vy1 && vx0) ? wy * (1.f - wx) * msk : 0.f;
      pw[wv][p][3] = (vy1 && vx1) ? wy * wx * msk : 0.f;
    }
    // prefetch next tap's offset/mask rows while gather+WMMA run
    if (tap < 8 && lane < 16) {
      const int pix = h * WC + w0 + lane;
      __builtin_prefetch(&om[((b * CMOUT + 2 * tap + 2) * HWC) + pix], 0, 3);
      __builtin_prefetch(&om[((b * CMOUT + 19 + tap) * HWC) + pix], 0, 3);
    }
    __syncthreads();
    // gather A tile: 16 pixels x 64 channels
    for (int i = lane; i < 16 * 64; i += 32) {
      const int p = i & 15, c = i >> 4;
      const float* xp = x + (size_t)(b * CINC + c) * HWC;
      const float v = pw[wv][p][0] * xp[pi[wv][p][0]]
                    + pw[wv][p][1] * xp[pi[wv][p][1]]
                    + pw[wv][p][2] * xp[pi[wv][p][2]]
                    + pw[wv][p][3] * xp[pi[wv][p][3]];
      At[wv][p][c] = f2bf(v);
    }
    __syncthreads();
    #pragma unroll
    for (int ks = 0; ks < 2; ++ks) {
      const int ca = ks * 32 + half * 8;
      v8bf aLo = *(const v8bf*)&At[wv][mn][ca];
      v8bf aHi = *(const v8bf*)&At[wv][mn][ca + 16];
      v16bf av = cat16(aLo, aHi);
      const int cbb = ks * 32 + half * 8;
      #pragma unroll
      for (int nt = 0; nt < 4; ++nt) {
        v8bf bl = *(const v8bf*)&Bt[nt * 16 + mn][cbb];
        v8bf bh = *(const v8bf*)&Bt[nt * 16 + mn][cbb + 16];
        acc[nt] = __builtin_amdgcn_wmma_f32_16x16x32_bf16(
            false, av, false, cat16(bl, bh), (short)0, acc[nt], false, false);
      }
    }
  }

  // epilogue: 8 consecutive pixels per lane per N-tile -> two b128 stores
  const int pixbase = h * WC + w0 + half * 8;
  #pragma unroll
  for (int nt = 0; nt < 4; ++nt) {
    const int o = nt * 16 + mn;
    const float bb = bias[o];
    float4 lo = make_float4(acc[nt][0] + bb, acc[nt][1] + bb,
                            acc[nt][2] + bb, acc[nt][3] + bb);
    float4 hi = make_float4(acc[nt][4] + bb, acc[nt][5] + bb,
                            acc[nt][6] + bb, acc[nt][7] + bb);
    float* dst = &out[((b * COUTC + o) * HWC) + pixbase];
    *(float4*)(dst)     = lo;
    *(float4*)(dst + 4) = hi;
  }
}

// ===================================================================
extern "C" void kernel_launch(void* const* d_in, const int* in_sizes, int n_in,
                              void* d_out, int out_size, void* d_ws, size_t ws_size,
                              hipStream_t stream) {
  (void)in_sizes; (void)n_in; (void)out_size; (void)ws_size;
  const float* x      = (const float*)d_in[0];
  const float* weight = (const float*)d_in[1];
  const float* bias   = (const float*)d_in[2];
  const float* cm_w   = (const float*)d_in[3];
  const float* cm_b   = (const float*)d_in[4];
  float* out = (float*)d_out;

  float* sim = (float*)d_ws;                              // B*8*H*W  = 2.62 MB
  float* om  = sim + (size_t)BB * 8 * HWC;                // B*27*H*W = 8.85 MB

  sim_kernel<<<(BB * HWC) / 256, 256, 0, stream>>>(x, sim);
  cm_conv_kernel<<<(BB * TILES_PER_B) / 4, 128, 0, stream>>>(x, sim, cm_w, cm_b, om);
  dcn_kernel<<<(BB * TILES_PER_B) / 4, 128, 0, stream>>>(x, om, weight, bias, out);
}
